// BERT_CRF_86371792323277
// MI455X (gfx1250) — compile-verified, matrix operand order checked by reading, not run on hardware
//
#include <hip/hip_runtime.h>
#include <hip/hip_bf16.h>

// ---------------------------------------------------------------------------
// BERT encoder (L=2) + CRF NLL for gfx1250 (CDNA5, wave32, WMMA).
// GEMMs AND attention run on v_wmma_f32_16x16x32_f16.
// ---------------------------------------------------------------------------

#define BB 16
#define SS 512
#define DD 768
#define LL 2
#define HH 12
#define TT 9
#define DH 64
#define FF 3072
#define QKVD (3 * DD)
#define ROWS (BB * SS)          // 8192
#define EPSLN 1e-12f

typedef __attribute__((ext_vector_type(16))) _Float16 v16h;
typedef __attribute__((ext_vector_type(8)))  _Float16 v8h;
typedef __attribute__((ext_vector_type(8)))  float    v8f;

__device__ __forceinline__ v16h make_frag(const _Float16* p0, const _Float16* p1) {
    v8h x0 = *(const v8h*)p0;
    v8h x1 = *(const v8h*)p1;
    v16h f;
    #pragma unroll
    for (int j = 0; j < 8; ++j) { f[j] = x0[j]; f[8 + j] = x1[j]; }
    return f;
}

// ---------------------------------------------------------------------------
// Weight transpose + f32->f16 convert:  in[K][N] (row-major) -> out[N][K]
// ---------------------------------------------------------------------------
__global__ void transpose_cvt_kernel(const float* __restrict__ in,
                                     _Float16* __restrict__ out,
                                     int K, int N) {
    size_t idx = (size_t)blockIdx.x * blockDim.x + threadIdx.x;
    size_t total = (size_t)K * N;
    if (idx >= total) return;
    int k = (int)(idx / N);
    int n = (int)(idx % N);
    out[(size_t)n * K + k] = (_Float16)in[idx];
}

// ---------------------------------------------------------------------------
// Register-blocked WMMA GEMM:
//   C[M][N] = A16[M][K] @ Bt16[N][K]^T + bias[N], optional exact GELU.
// Block = 256 threads = 8 waves (2 waveM x 4 waveN). Wave tile = 64x64
// (4x4 WMMA tiles). Block tile = 128x256. M%128==0, N%256==0, K%32==0.
// ---------------------------------------------------------------------------
__global__ void __launch_bounds__(256, 1)
gemm_wmma_kernel(const _Float16* __restrict__ A,
                 const _Float16* __restrict__ Bt,
                 const float* __restrict__ bias,
                 float* __restrict__ C,        // nullable
                 _Float16* __restrict__ C16,   // nullable
                 int M, int N, int K, int gelu) {
    const int lane  = threadIdx.x & 31;
    const int wave  = threadIdx.x >> 5;
    const int waveM = wave >> 2;        // 0..1
    const int waveN = wave & 3;         // 0..3
    const int tileM = blockIdx.y * 128 + waveM * 64;
    const int tileN = blockIdx.x * 256 + waveN * 64;
    const int half  = lane >> 4;        // K-half selector (ISA 16-bit A layout)
    const int r15   = lane & 15;

    const _Float16* ap = A  + (size_t)(tileM + r15) * K + half * 8;
    const _Float16* bp = Bt + (size_t)(tileN + r15) * K + half * 8;
    const size_t rstep = (size_t)16 * K;      // advance 16 rows

    v8f acc[4][4] = {};
    for (int k0 = 0; k0 < K; k0 += 32) {
        v16h afr[4], bfr[4];
        #pragma unroll
        for (int i = 0; i < 4; ++i) {
            const _Float16* api = ap + (size_t)i * rstep + k0;
            const _Float16* bpi = bp + (size_t)i * rstep + k0;
            afr[i] = make_frag(api, api + 16);
            bfr[i] = make_frag(bpi, bpi + 16);
        }
        if (k0 + 32 < K) {   // L2 prefetch of next K-block
            __builtin_prefetch(ap + k0 + 32, 0, 1);
            __builtin_prefetch(bp + k0 + 32, 0, 1);
        }
        #pragma unroll
        for (int i = 0; i < 4; ++i) {
            #pragma unroll
            for (int j = 0; j < 4; ++j) {
                acc[i][j] = __builtin_amdgcn_wmma_f32_16x16x32_f16(
                    false, afr[i], false, bfr[j],
                    (short)0, acc[i][j], false, false);
            }
        }
    }

    // Epilogue: bias (+ GELU), store f32 and/or f16.
    #pragma unroll
    for (int i = 0; i < 4; ++i) {
        const int mrow = tileM + i * 16 + half * 8;
        #pragma unroll
        for (int j = 0; j < 4; ++j) {
            const int n   = tileN + j * 16 + r15;
            const float b = bias ? bias[n] : 0.0f;
            #pragma unroll
            for (int r = 0; r < 8; ++r) {
                float val = acc[i][j][r] + b;
                if (gelu) val = 0.5f * val * (1.0f + erff(val * 0.70710678118654752f));
                size_t idx = (size_t)(mrow + r) * N + n;
                if (C)   C[idx]   = val;
                if (C16) C16[idx] = (_Float16)val;
            }
        }
    }
}

// ---------------------------------------------------------------------------
// Repack qkv f32 [row][3D] -> q16/k16 [b,h,s,64] f16 and vT16 [b,h,d,s] f16.
// ---------------------------------------------------------------------------
__global__ void repack_qkv_kernel(const float* __restrict__ qkv,
                                  _Float16* __restrict__ q16,
                                  _Float16* __restrict__ k16,
                                  _Float16* __restrict__ vT16) {
    size_t idx = (size_t)blockIdx.x * blockDim.x + threadIdx.x;
    if (idx >= (size_t)ROWS * DD) return;
    int row = (int)(idx / DD);
    int c   = (int)(idx % DD);          // c = h*64 + d
    int b = row >> 9, s = row & (SS - 1);
    int h = c >> 6,  d = c & (DH - 1);
    const float* src = qkv + (size_t)row * QKVD;
    size_t qk = (((size_t)b * HH + h) * SS + s) * DH + d;
    q16[qk] = (_Float16)src[c];
    k16[qk] = (_Float16)src[DD + c];
    vT16[(((size_t)b * HH + h) * DH + d) * SS + s] = (_Float16)src[2 * DD + c];
}

// ---------------------------------------------------------------------------
// WMMA attention: one wave per (16-query tile, head, batch).
// Pass1: scores(16x512) = q(16x64) @ K^T via WMMA -> LDS (scaled 1/8).
// Softmax rows with wave shuffles, exp -> f16 in LDS.
// Pass2: ctx(16x64) = P(16x512) @ V via WMMA (A from LDS, B from vT16).
// ---------------------------------------------------------------------------
__global__ void __launch_bounds__(32, 1)
attention_wmma_kernel(const _Float16* __restrict__ q16,
                      const _Float16* __restrict__ k16,
                      const _Float16* __restrict__ vT16,
                      _Float16* __restrict__ ctx16) {
    const int qt   = blockIdx.x;        // 0..31 (query tile)
    const int hh   = blockIdx.y;
    const int b    = blockIdx.z;
    const int lane = threadIdx.x & 31;
    const int r15  = lane & 15;
    const int half = lane >> 4;

    __shared__ float    sc[16 * SS];    // 32 KB scores
    __shared__ _Float16 pp[16 * SS];    // 16 KB probabilities (f16)
    __shared__ float    rs[16];         // reciprocal row sums

    const size_t head = (size_t)b * HH + hh;

    // Q fragments: rows qt*16 + (0..15), dims 0..63 (two K=32 chunks)
    const _Float16* qp = q16 + (head * SS + qt * 16 + r15) * DH + half * 8;
    const v16h aq0 = make_frag(qp,      qp + 16);
    const v16h aq1 = make_frag(qp + 32, qp + 48);

    // ---- Pass 1: scores ----
    for (int kt = 0; kt < SS / 16; ++kt) {
        const _Float16* kp = k16 + (head * SS + kt * 16 + r15) * DH + half * 8;
        v16h bk0 = make_frag(kp,      kp + 16);
        v16h bk1 = make_frag(kp + 32, kp + 48);
        v8f s = {};
        s = __builtin_amdgcn_wmma_f32_16x16x32_f16(false, aq0, false, bk0,
                                                   (short)0, s, false, false);
        s = __builtin_amdgcn_wmma_f32_16x16x32_f16(false, aq1, false, bk1,
                                                   (short)0, s, false, false);
        #pragma unroll
        for (int r = 0; r < 8; ++r)
            sc[(r + half * 8) * SS + kt * 16 + r15] = s[r] * 0.125f;
    }
    __syncthreads();

    // ---- Softmax (16 rows, whole wave per row) ----
    for (int r = 0; r < 16; ++r) {
        float mx = -1e30f;
        for (int c = lane; c < SS; c += 32) mx = fmaxf(mx, sc[r * SS + c]);
        #pragma unroll
        for (int off = 16; off; off >>= 1) mx = fmaxf(mx, __shfl_xor(mx, off, 32));
        float sum = 0.f;
        for (int c = lane; c < SS; c += 32) {
            float e = __expf(sc[r * SS + c] - mx);
            pp[r * SS + c] = (_Float16)e;
            sum += e;
        }
        #pragma unroll
        for (int off = 16; off; off >>= 1) sum += __shfl_xor(sum, off, 32);
        if (lane == 0) rs[r] = 1.0f / sum;
    }
    __syncthreads();

    // ---- Pass 2: ctx = P @ V ----
    v8f acc[4] = {};
    const _Float16* vbase = vT16 + head * DH * SS;
    for (int kt = 0; kt < SS; kt += 32) {
        const _Float16* pa = &pp[r15 * SS + kt + half * 8];
        v16h ap = make_frag(pa, pa + 16);
        #pragma unroll
        for (int nt = 0; nt < 4; ++nt) {
            const _Float16* vp = vbase + (size_t)(nt * 16 + r15) * SS + kt + half * 8;
            v16h bv = make_frag(vp, vp + 16);
            acc[nt] = __builtin_amdgcn_wmma_f32_16x16x32_f16(
                false, ap, false, bv, (short)0, acc[nt], false, false);
        }
    }

    // ---- Store ctx16[b, q, hh*64 + col] ----
    #pragma unroll
    for (int nt = 0; nt < 4; ++nt) {
        #pragma unroll
        for (int r = 0; r < 8; ++r) {
            int rowq = qt * 16 + r + half * 8;
            int col  = nt * 16 + r15;
            float v  = acc[nt][r] * rs[r + half * 8];
            ctx16[((size_t)b * SS + rowq) * DD + hh * DH + col] = (_Float16)v;
        }
    }
}

// ---------------------------------------------------------------------------
// Embedding lookup + LayerNorm. One block (256 thr) per row; 3 elems/thread.
// ---------------------------------------------------------------------------
__global__ void embed_ln_kernel(const int* __restrict__ x,
                                const float* __restrict__ wemb,
                                const float* __restrict__ pemb,
                                const float* __restrict__ gs,
                                const float* __restrict__ gb,
                                float* __restrict__ out,
                                _Float16* __restrict__ out16) {
    const int row = blockIdx.x;
    const int s   = row & (SS - 1);
    const int tok = x[row];
    const int t   = threadIdx.x;
    __shared__ float r1[256], r2[256];
    float v[3], s1 = 0.f, s2 = 0.f;
    #pragma unroll
    for (int j = 0; j < 3; ++j) {
        int d = t + j * 256;
        float val = wemb[(size_t)tok * DD + d] + pemb[(size_t)s * DD + d];
        v[j] = val; s1 += val; s2 += val * val;
    }
    r1[t] = s1; r2[t] = s2; __syncthreads();
    for (int off = 128; off; off >>= 1) {
        if (t < off) { r1[t] += r1[t + off]; r2[t] += r2[t + off]; }
        __syncthreads();
    }
    const float mean = r1[0] * (1.0f / DD);
    const float var  = r2[0] * (1.0f / DD) - mean * mean;
    const float rstd = rsqrtf(var + EPSLN);
    #pragma unroll
    for (int j = 0; j < 3; ++j) {
        int d = t + j * 256;
        float o = (v[j] - mean) * rstd * gs[d] + gb[d];
        size_t idx = (size_t)row * DD + d;
        out[idx] = o; out16[idx] = (_Float16)o;
    }
}

// ---------------------------------------------------------------------------
// out = LN(xin + resid) * gs + gb ; also f16 copy. Block per row.
// ---------------------------------------------------------------------------
__global__ void add_ln_kernel(const float* __restrict__ xin,
                              const float* __restrict__ resid,
                              const float* __restrict__ gs,
                              const float* __restrict__ gb,
                              float* __restrict__ out,
                              _Float16* __restrict__ out16) {
    const int row = blockIdx.x;
    const int t   = threadIdx.x;
    __shared__ float r1[256], r2[256];
    float v[3], s1 = 0.f, s2 = 0.f;
    #pragma unroll
    for (int j = 0; j < 3; ++j) {
        int d = t + j * 256;
        size_t idx = (size_t)row * DD + d;
        float val = xin[idx] + resid[idx];
        v[j] = val; s1 += val; s2 += val * val;
    }
    r1[t] = s1; r2[t] = s2; __syncthreads();
    for (int off = 128; off; off >>= 1) {
        if (t < off) { r1[t] += r1[t + off]; r2[t] += r2[t + off]; }
        __syncthreads();
    }
    const float mean = r1[0] * (1.0f / DD);
    const float var  = r2[0] * (1.0f / DD) - mean * mean;
    const float rstd = rsqrtf(var + EPSLN);
    #pragma unroll
    for (int j = 0; j < 3; ++j) {
        int d = t + j * 256;
        float o = (v[j] - mean) * rstd * gs[d] + gb[d];
        size_t idx = (size_t)row * DD + d;
        out[idx] = o; out16[idx] = (_Float16)o;
    }
}

// ---------------------------------------------------------------------------
// em[row][t] = h[row] . W_tag[:,t] + b_tag[t]   (T=9; one wave per row)
// ---------------------------------------------------------------------------
__global__ void tag_proj_kernel(const float* __restrict__ h,
                                const float* __restrict__ Wt,
                                const float* __restrict__ bt,
                                float* __restrict__ em) {
    const int row = blockIdx.x;
    const int t   = threadIdx.x;
    if (t >= TT) return;
    float s = bt[t];
    const float* hp = h + (size_t)row * DD;
    for (int d = 0; d < DD; ++d) s += hp[d] * Wt[(size_t)d * TT + t];
    em[(size_t)row * TT + t] = s;
}

__global__ void zero_kernel(float* p) { p[0] = 0.0f; }

// ---------------------------------------------------------------------------
// CRF negative log-likelihood. Block per batch element, 32 threads.
// ---------------------------------------------------------------------------
__global__ void crf_kernel(const float* __restrict__ em,
                           const int* __restrict__ y,
                           const float* __restrict__ start,
                           const float* __restrict__ endv,
                           const float* __restrict__ trans,
                           float* __restrict__ out) {
    const int b = blockIdx.x;
    const int t = threadIdx.x;
    const int NPOS = SS - 1;                       // 511
    const float* e  = em + (size_t)b * SS * TT + TT;   // skip position 0
    const int*   tg = y  + (size_t)b * SS + 1;

    __shared__ float al[TT];

    if (t < TT) al[t] = start[t] + e[t];
    __syncthreads();

    for (int i = 1; i < NPOS; ++i) {
        const bool mt = tg[i] > 0;
        float nxt = 0.f;
        if (t < TT) {
            float mx = -1e30f;
            #pragma unroll
            for (int p = 0; p < TT; ++p) mx = fmaxf(mx, al[p] + trans[p * TT + t]);
            float s = 0.f;
            #pragma unroll
            for (int p = 0; p < TT; ++p) s += __expf(al[p] + trans[p * TT + t] - mx);
            nxt = mx + __logf(s) + e[(size_t)i * TT + t];
        }
        __syncthreads();
        if (t < TT && mt) al[t] = nxt;
        __syncthreads();
    }

    if (t == 0) {
        int tg0 = tg[0];
        float num = start[tg0] + e[tg0];
        int cnt = (tg0 > 0) ? 1 : 0;
        for (int i = 1; i < NPOS; ++i) {
            int ti = tg[i];
            if (ti > 0) {
                num += trans[tg[i - 1] * TT + ti] + e[(size_t)i * TT + ti];
                ++cnt;
            }
        }
        int last = cnt - 1;
        num += endv[tg[last]];
        float mx = -1e30f;
        for (int p = 0; p < TT; ++p) mx = fmaxf(mx, al[p] + endv[p]);
        float s = 0.f;
        for (int p = 0; p < TT; ++p) s += __expf(al[p] + endv[p] - mx);
        float logZ = mx + __logf(s);
        atomicAdd(out, logZ - num);
    }
}

// ---------------------------------------------------------------------------
// Host orchestration
// ---------------------------------------------------------------------------
extern "C" void kernel_launch(void* const* d_in, const int* in_sizes, int n_in,
                              void* d_out, int out_size, void* d_ws, size_t ws_size,
                              hipStream_t stream) {
    (void)in_sizes; (void)n_in; (void)out_size; (void)ws_size;

    const int*   x      = (const int*)d_in[0];
    const int*   y      = (const int*)d_in[1];
    const float* wemb   = (const float*)d_in[2];
    const float* pemb   = (const float*)d_in[3];
    const float* ln_e_s = (const float*)d_in[4];
    const float* ln_e_b = (const float*)d_in[5];
    const float* Wqkv   = (const float*)d_in[6];
    const float* bqkv   = (const float*)d_in[7];
    const float* Wo     = (const float*)d_in[8];
    const float* bo     = (const float*)d_in[9];
    const float* ln1_s  = (const float*)d_in[10];
    const float* ln1_b  = (const float*)d_in[11];
    const float* W1     = (const float*)d_in[12];
    const float* b1     = (const float*)d_in[13];
    const float* W2     = (const float*)d_in[14];
    const float* b2     = (const float*)d_in[15];
    const float* ln2_s  = (const float*)d_in[16];
    const float* ln2_b  = (const float*)d_in[17];
    const float* W_tag  = (const float*)d_in[18];
    const float* b_tag  = (const float*)d_in[19];
    const float* cstart = (const float*)d_in[20];
    const float* cend   = (const float*)d_in[21];
    const float* ctrans = (const float*)d_in[22];
    float* out = (float*)d_out;

    // Carve workspace
    char* p = (char*)d_ws;
    auto carve = [&](size_t bytes) {
        void* r = (void*)p;
        p += (bytes + 255) & ~(size_t)255;
        return r;
    };
    float*    h      = (float*)   carve((size_t)ROWS * DD * 4);
    _Float16* h16    = (_Float16*)carve((size_t)ROWS * DD * 2);
    float*    qkv    = (float*)   carve((size_t)ROWS * QKVD * 4);
    _Float16* q16    = (_Float16*)carve((size_t)ROWS * DD * 2);
    _Float16* k16    = (_Float16*)carve((size_t)ROWS * DD * 2);
    _Float16* vT16   = (_Float16*)carve((size_t)ROWS * DD * 2);
    _Float16* ctx16  = (_Float16*)carve((size_t)ROWS * DD * 2);
    _Float16* ff16   = (_Float16*)carve((size_t)ROWS * FF * 2);
    float*    proj   = (float*)   carve((size_t)ROWS * DD * 4);
    float*    em     = (float*)   carve((size_t)ROWS * TT * 4);
    _Float16* WqkvT  = (_Float16*)carve((size_t)QKVD * DD * 2);
    _Float16* WoT    = (_Float16*)carve((size_t)DD * DD * 2);
    _Float16* W1T    = (_Float16*)carve((size_t)FF * DD * 2);
    _Float16* W2T    = (_Float16*)carve((size_t)DD * FF * 2);

    // Embedding + LN
    embed_ln_kernel<<<ROWS, 256, 0, stream>>>(x, wemb, pemb, ln_e_s, ln_e_b, h, h16);

    auto tgrid = [](size_t n) { return (unsigned)((n + 255) / 256); };

    for (int l = 0; l < LL; ++l) {
        const float* Wqkv_l = Wqkv + (size_t)l * DD * QKVD;
        const float* Wo_l   = Wo   + (size_t)l * DD * DD;
        const float* W1_l   = W1   + (size_t)l * DD * FF;
        const float* W2_l   = W2   + (size_t)l * FF * DD;

        // ---- QKV projection ----
        transpose_cvt_kernel<<<tgrid((size_t)DD * QKVD), 256, 0, stream>>>(Wqkv_l, WqkvT, DD, QKVD);
        gemm_wmma_kernel<<<dim3(QKVD / 256, ROWS / 128), 256, 0, stream>>>(
            h16, WqkvT, bqkv + (size_t)l * QKVD, qkv, nullptr, ROWS, QKVD, DD, 0);

        // ---- Attention (WMMA) ----
        repack_qkv_kernel<<<tgrid((size_t)ROWS * DD), 256, 0, stream>>>(qkv, q16, k16, vT16);
        attention_wmma_kernel<<<dim3(SS / 16, HH, BB), 32, 0, stream>>>(q16, k16, vT16, ctx16);

        // ---- Output projection + residual + LN ----
        transpose_cvt_kernel<<<tgrid((size_t)DD * DD), 256, 0, stream>>>(Wo_l, WoT, DD, DD);
        gemm_wmma_kernel<<<dim3(DD / 256, ROWS / 128), 256, 0, stream>>>(
            ctx16, WoT, bo + (size_t)l * DD, proj, nullptr, ROWS, DD, DD, 0);
        add_ln_kernel<<<ROWS, 256, 0, stream>>>(
            proj, h, ln1_s + (size_t)l * DD, ln1_b + (size_t)l * DD, h, h16);

        // ---- FFN: GELU(h W1 + b1) W2 + b2, residual + LN ----
        transpose_cvt_kernel<<<tgrid((size_t)DD * FF), 256, 0, stream>>>(W1_l, W1T, DD, FF);
        gemm_wmma_kernel<<<dim3(FF / 256, ROWS / 128), 256, 0, stream>>>(
            h16, W1T, b1 + (size_t)l * FF, nullptr, ff16, ROWS, FF, DD, 1);
        transpose_cvt_kernel<<<tgrid((size_t)FF * DD), 256, 0, stream>>>(W2_l, W2T, FF, DD);
        gemm_wmma_kernel<<<dim3(DD / 256, ROWS / 128), 256, 0, stream>>>(
            ff16, W2T, b2 + (size_t)l * DD, proj, nullptr, ROWS, DD, FF, 0);
        add_ln_kernel<<<ROWS, 256, 0, stream>>>(
            proj, h, ln2_s + (size_t)l * DD, ln2_b + (size_t)l * DD, h, h16);
    }

    // ---- Tag projection (T=9) ----
    tag_proj_kernel<<<ROWS, 32, 0, stream>>>(h, W_tag, b_tag, em);

    // ---- CRF NLL ----
    zero_kernel<<<1, 1, 0, stream>>>(out);
    crf_kernel<<<BB, 32, 0, stream>>>(em, y, cstart, cend, ctrans, out);
}
